// Mamba2DecoderLayer_23622320128270
// MI455X (gfx1250) — matrix operand order repaired and checked by reading, not compiled, not run on hardware
//
#include <hip/hip_runtime.h>
#include <cstdint>
#include <cstddef>

// ---------------- problem constants ----------------
#define D_MODEL   4096
#define D_XB      1024
#define D_INNER   4096
#define NHEADS    32
#define HEADDIM   128
#define DSTATE    128
#define DCONV     4
#define INTER     14336
#define BB        2
#define LL        2048
#define TOK       (BB*LL)                         // 4096 tokens
#define PROJ_OUT  (2*D_INNER + 2*D_XB + NHEADS)   // 10272
#define CONV_DIM  (2*D_XB + D_INNER)              // 6144
#define CHUNK     128
#define NCHUNK    (LL/CHUNK)                      // 16
#define EPSF      1e-5f

typedef __attribute__((ext_vector_type(16))) __bf16 v16bf;
typedef __attribute__((ext_vector_type(8)))  float  v8f;

union FragU  { uint4 u4[2]; v16bf v; };
union H16U   { __bf16 h[16]; uint4 u4[2]; };

// LDS byte address (DS-style offset) of a generic pointer into __shared__.
__device__ __forceinline__ uint32_t lds_addr_u32(const void* p) {
  return (uint32_t)(uintptr_t)(__attribute__((address_space(3))) const void*)p;
}

// ---------------- WMMA helpers ----------------
__device__ __forceinline__ v8f wmma_bf16(v16bf a, v16bf b, v8f c) {
  // D = A(16x32 bf16) * B(32x16 bf16) + C(16x16 f32)
  return __builtin_amdgcn_wmma_f32_16x16x32_bf16(false, a, false, b,
                                                 (short)0, c, false, false);
}

// A-fragment: row-major bf16 tile in LDS, stride in halves.
// A layout: lane m=L%16, hi=L/16; elems are 16 contiguous k at kbase+hi*16
// (matches B-fragment k order).
__device__ __forceinline__ v16bf lda_bf(const __bf16* s, int stride,
                                        int mbase, int kbase) {
  int lane = threadIdx.x & 31;
  int hi   = (lane >> 4) & 1;
  const __bf16* p = s + (size_t)(mbase + (lane & 15)) * stride + kbase;
  FragU f;
  f.u4[0] = *(const uint4*)(p + hi * 16);
  f.u4[1] = *(const uint4*)(p + hi * 16 + 8);
  return f.v;
}

// B-fragment from W-form tile (rows = output col n, inner dim = k).
// Lane n=L%16, hi=L/16; elems are 16 contiguous k at kbase+hi*16.
__device__ __forceinline__ v16bf ldb_bf(const __bf16* s, int stride,
                                        int nbase, int kbase) {
  int lane = threadIdx.x & 31;
  int hi   = (lane >> 4) & 1;
  const __bf16* p = s + (size_t)(nbase + (lane & 15)) * stride + kbase + hi * 16;
  FragU f;
  f.u4[0] = *(const uint4*)(p);
  f.u4[1] = *(const uint4*)(p + 8);
  return f.v;
}

// B-fragment sourced from an f32 LDS tile (the SSD state S), cvt to bf16.
__device__ __forceinline__ v16bf ldb_f32(const float* s, int stride,
                                         int nbase, int kbase) {
  int lane = threadIdx.x & 31;
  int hi   = (lane >> 4) & 1;
  const float* p = s + (size_t)(nbase + (lane & 15)) * stride + kbase + hi * 16;
  v16bf r;
#pragma unroll
  for (int e = 0; e < 16; ++e) r[e] = (__bf16)p[e];
  return r;
}

__device__ __forceinline__ float siluf(float x) {
  return x / (1.f + __expf(-x));
}

// ---------------- RMSNorm -> bf16 ----------------
__global__ __launch_bounds__(256)
void rmsnorm_kernel(const float* __restrict__ x, const float* __restrict__ w,
                    __bf16* __restrict__ out, int D) {
  __shared__ float red[256];
  size_t row = blockIdx.x;
  const float* xr = x + row * D;
  int tid = threadIdx.x;
  float s = 0.f;
  for (int i = tid; i < D; i += 256) { float v = xr[i]; s += v * v; }
  red[tid] = s;
  __syncthreads();
  for (int off = 128; off > 0; off >>= 1) {
    if (tid < off) red[tid] += red[tid + off];
    __syncthreads();
  }
  float sc = rsqrtf(red[0] / (float)D + EPSF);
  for (int i = tid; i < D; i += 256)
    out[row * D + i] = (__bf16)(xr[i] * sc * w[i]);
}

// ---------------- generic WMMA GEMM (double-buffered, async A staging) ----
// C[M,N] = A[M,K](bf16,row-major) * W[N,K](f32,row-major)^T
// epi: 0 -> outF = acc
//      1 -> outF = acc + res
//      2 -> outB = bf16(acc)
//      3 -> outB = bf16(silu(acc) * mul)        (fused gate*up)
__global__ __launch_bounds__(256)
void gemm_kernel(const __bf16* __restrict__ A, const float* __restrict__ W,
                 int M, int N, int K,
                 float* __restrict__ outF, __bf16* __restrict__ outB,
                 const float* __restrict__ res, const __bf16* __restrict__ mul,
                 int epi) {
  __shared__ __bf16 sA[2][128 * 40];   // 128 x 32, stride 40 halves (80B rows)
  __shared__ __bf16 sB[2][128 * 40];

  int m0 = blockIdx.x * 128;
  int n0 = blockIdx.y * 128;
  int tid = threadIdx.x;
  int lane = tid & 31;
  int wid = tid >> 5;
  int wm = (wid & 3) * 32;   // wave M offset within tile
  int wn = (wid >> 2) * 64;  // wave N offset within tile

  v8f acc[2][4];
#pragma unroll
  for (int tm = 0; tm < 2; ++tm)
#pragma unroll
    for (int tn = 0; tn < 4; ++tn) acc[tm][tn] = (v8f){};

  int row = tid >> 1;
  int seg = (tid & 1) * 16;          // in elements (halves / floats)
  int nrow = n0 + row;

  // per-thread LDS byte addresses of the A-tile slots (both buffers)
  uint32_t ldsA0 = lds_addr_u32(&sA[0][row * 40 + seg]);
  uint32_t ldsA1 = lds_addr_u32(&sA[1][row * 40 + seg]);

  auto stage = [&](int buf, int k0) {
    // ---- A tile: async DMA global -> LDS (bf16, no conversion needed) ----
    uint64_t ga = (uint64_t)(uintptr_t)(A + (size_t)(m0 + row) * K + k0 + seg);
    uint32_t la = buf ? ldsA1 : ldsA0;
    asm volatile(
        "global_load_async_to_lds_b128 %0, %1, off\n\t"
        "global_load_async_to_lds_b128 %0, %1, off offset:16"
        :: "v"(la), "v"(ga) : "memory");
    // ---- W tile: f32 -> bf16 in VALU, store to LDS ----
    H16U hb;
    if (nrow < N) {
      const float* wp = W + (size_t)nrow * K + k0 + seg;
#pragma unroll
      for (int j = 0; j < 4; ++j) {
        float4 t = *(const float4*)(wp + j * 4);
        hb.h[j * 4 + 0] = (__bf16)t.x;
        hb.h[j * 4 + 1] = (__bf16)t.y;
        hb.h[j * 4 + 2] = (__bf16)t.z;
        hb.h[j * 4 + 3] = (__bf16)t.w;
      }
      if (k0 + 32 < K)  // pull next weight slab toward L2
        __builtin_prefetch(wp + 32, 0, 0);
    } else {
#pragma unroll
      for (int j = 0; j < 16; ++j) hb.h[j] = (__bf16)0.f;
    }
    *(uint4*)(&sB[buf][row * 40 + seg])     = hb.u4[0];
    *(uint4*)(&sB[buf][row * 40 + seg + 8]) = hb.u4[1];
  };

  int nk = K >> 5;
  stage(0, 0);
  int buf = 0;
  for (int it = 0; it < nk; ++it) {
    // make async A-copies of the buffer we are about to read visible,
    // then publish the tile (barrier also orders the ds_store W writes)
    asm volatile("s_wait_asynccnt 0x0" ::: "memory");
    __syncthreads();
    if (it + 1 < nk) stage(buf ^ 1, (it + 1) * 32);

#pragma unroll
    for (int tm = 0; tm < 2; ++tm) {
      v16bf af = lda_bf(sA[buf], 40, wm + tm * 16, 0);
#pragma unroll
      for (int tn = 0; tn < 4; ++tn) {
        v16bf bf = ldb_bf(sB[buf], 40, wn + tn * 16, 0);
        acc[tm][tn] = wmma_bf16(af, bf, acc[tm][tn]);
      }
    }
    buf ^= 1;
  }

  int hi = (lane >> 4) & 1;
#pragma unroll
  for (int tm = 0; tm < 2; ++tm)
#pragma unroll
    for (int tn = 0; tn < 4; ++tn)
#pragma unroll
      for (int i = 0; i < 8; ++i) {
        int gm = m0 + wm + tm * 16 + hi * 8 + i;
        int gn = n0 + wn + tn * 16 + (lane & 15);
        if (gn < N) {
          size_t o = (size_t)gm * N + gn;
          float v = acc[tm][tn][i];
          if (epi == 0)      outF[o] = v;
          else if (epi == 1) outF[o] = v + res[o];
          else if (epi == 2) outB[o] = (__bf16)v;
          else               outB[o] = (__bf16)(siluf(v) * (float)mul[o]);
        }
      }
}

// ---------------- depthwise causal conv + SiLU ----------------
__global__ __launch_bounds__(256)
void conv_kernel(const float* __restrict__ zx, const float* __restrict__ cw,
                 const float* __restrict__ cb, float* __restrict__ xconv) {
  int idx = blockIdx.x * 256 + threadIdx.x;
  if (idx >= TOK * CONV_DIM) return;
  int ch  = idx % CONV_DIM;
  int pos = idx / CONV_DIM;       // b*LL + l
  int l   = pos % LL;
  float acc = cb[ch];
#pragma unroll
  for (int t = 0; t < DCONV; ++t) {
    int ls = l - (DCONV - 1) + t;
    if (ls >= 0) {
      float xin = zx[(size_t)(pos - (l - ls)) * PROJ_OUT + D_INNER + ch];
      acc += cw[ch * DCONV + t] * xin;
    }
  }
  xconv[(size_t)pos * CONV_DIM + ch] = siluf(acc);
}

// ---------------- dt = softplus(dt_raw + bias) ----------------
__global__ __launch_bounds__(256)
void dt_kernel(const float* __restrict__ zx, const float* __restrict__ dt_bias,
               float* __restrict__ dtbuf) {
  int idx = blockIdx.x * 256 + threadIdx.x;
  if (idx >= TOK * NHEADS) return;
  int h = idx % NHEADS;
  int pos = idx / NHEADS;
  float x = zx[(size_t)pos * PROJ_OUT + (PROJ_OUT - NHEADS) + h] + dt_bias[h];
  float sp = (x > 20.f) ? x : log1pf(__expf(x));
  dtbuf[(size_t)pos * NHEADS + h] = sp;
}

// ---------------- chunked SSD scan (WMMA) ----------------
// one workgroup per (b,h); sequential over NCHUNK chunks keeping S in LDS.
#define SSD_STRB 136   // bf16 tile stride (halves)
#define SSD_STRS 132   // f32 state stride (floats)
#define SSD_LDS_BYTES (5*128*SSD_STRB*2 + 128*SSD_STRS*4 + 128*4)

__global__ __launch_bounds__(256)
void ssd_kernel(const float* __restrict__ xconv, const float* __restrict__ dtbuf,
                const float* __restrict__ zx, const float* __restrict__ A_log,
                const float* __restrict__ Dp, __bf16* __restrict__ yact) {
  extern __shared__ char smem[];
  __bf16* sCc  = (__bf16*)smem;                 // [q][n]
  __bf16* sBc  = sCc  + 128 * SSD_STRB;         // [q][n]
  __bf16* sBcT = sBc  + 128 * SSD_STRB;         // [n][q] * dec_end(q)
  __bf16* sXT  = sBcT + 128 * SSD_STRB;         // [p][q] = x*dt transposed
  __bf16* sAtt = sXT  + 128 * SSD_STRB;         // [q][k]
  float*  sS   = (float*)(sAtt + 128 * SSD_STRB);  // state [p][n]
  float*  cs   = sS + 128 * SSD_STRS;           // cumsum

  int b = blockIdx.x >> 5;
  int h = blockIdx.x & 31;
  int g = h >> 2;                 // x/B group (rep = 4)
  int tid = threadIdx.x, lane = tid & 31, wid = tid >> 5;
  int wm = (wid & 3) * 32, wn = (wid >> 2) * 64;
  int hi = (lane >> 4) & 1;

  float Aneg = -__expf(A_log[h]);
  float Dh = Dp[h];

  for (int i = tid; i < 128 * SSD_STRS; i += 256) sS[i] = 0.f;
  __syncthreads();

  for (int c = 0; c < NCHUNK; ++c) {
    size_t tok0 = (size_t)b * LL + c * CHUNK;

    // ---- 1. a = dt*A, inclusive cumsum over the chunk ----
    if (tid < 128)
      cs[tid] = dtbuf[(tok0 + tid) * NHEADS + h] * Aneg;
    __syncthreads();
    for (int off = 1; off < 128; off <<= 1) {
      float v = 0.f;
      if (tid < 128 && tid >= off) v = cs[tid - off];
      __syncthreads();
      if (tid < 128) cs[tid] += v;
      __syncthreads();
    }
    float csl = cs[127];
    float gch = __expf(csl);

    // ---- 2. stage tiles (f32 -> bf16) ----
    for (int i = tid; i < 128 * 128; i += 256) {
      int q = i >> 7, n = i & 127;
      size_t ro = (tok0 + q) * CONV_DIM;
      float cv  = xconv[ro + 2 * D_XB + h * DSTATE + n];
      float bv  = xconv[ro + D_XB + g * DSTATE + n];
      float xv  = xconv[ro + g * HEADDIM + n];         // n plays p here
      float dtv = dtbuf[(tok0 + q) * NHEADS + h];
      float de  = __expf(csl - cs[q]);                 // dec_end
      sCc [q * SSD_STRB + n] = (__bf16)cv;
      sBc [q * SSD_STRB + n] = (__bf16)bv;
      sBcT[n * SSD_STRB + q] = (__bf16)(bv * de);
      sXT [n * SSD_STRB + q] = (__bf16)(xv * dtv);
    }
    __syncthreads();

    // ---- 3. att = (Cc . Bc^T) * Ldec, bf16 into sAtt ----
    {
      v8f a1[2][4];
#pragma unroll
      for (int tm = 0; tm < 2; ++tm)
#pragma unroll
        for (int tn = 0; tn < 4; ++tn) a1[tm][tn] = (v8f){};
      for (int ks = 0; ks < 128; ks += 32)
#pragma unroll
        for (int tm = 0; tm < 2; ++tm) {
          v16bf af = lda_bf(sCc, SSD_STRB, wm + tm * 16, ks);
#pragma unroll
          for (int tn = 0; tn < 4; ++tn)
            a1[tm][tn] = wmma_bf16(af, ldb_bf(sBc, SSD_STRB, wn + tn * 16, ks),
                                   a1[tm][tn]);
        }
#pragma unroll
      for (int tm = 0; tm < 2; ++tm)
#pragma unroll
        for (int tn = 0; tn < 4; ++tn)
#pragma unroll
          for (int i = 0; i < 8; ++i) {
            int q = wm + tm * 16 + hi * 8 + i;
            int k = wn + tn * 16 + (lane & 15);
            float v = (q >= k) ? a1[tm][tn][i] * __expf(cs[q] - cs[k]) : 0.f;
            sAtt[q * SSD_STRB + k] = (__bf16)v;
          }
    }
    __syncthreads();

    // ---- 4. Yd = att.XdtT^T ; Yo = Cc.S^T ; Snew = XdtT.BcT^T ----
    v8f aY[2][4], aYo[2][4], aS[2][4];
#pragma unroll
    for (int tm = 0; tm < 2; ++tm)
#pragma unroll
      for (int tn = 0; tn < 4; ++tn) {
        aY[tm][tn] = (v8f){}; aYo[tm][tn] = (v8f){}; aS[tm][tn] = (v8f){};
      }
    for (int ks = 0; ks < 128; ks += 32) {
#pragma unroll
      for (int tm = 0; tm < 2; ++tm) {
        v16bf af = lda_bf(sAtt, SSD_STRB, wm + tm * 16, ks);
#pragma unroll
        for (int tn = 0; tn < 4; ++tn)
          aY[tm][tn] = wmma_bf16(af, ldb_bf(sXT, SSD_STRB, wn + tn * 16, ks),
                                 aY[tm][tn]);
      }
#pragma unroll
      for (int tm = 0; tm < 2; ++tm) {
        v16bf af = lda_bf(sCc, SSD_STRB, wm + tm * 16, ks);
#pragma unroll
        for (int tn = 0; tn < 4; ++tn)
          aYo[tm][tn] = wmma_bf16(af, ldb_f32(sS, SSD_STRS, wn + tn * 16, ks),
                                  aYo[tm][tn]);
      }
#pragma unroll
      for (int tm = 0; tm < 2; ++tm) {
        v16bf af = lda_bf(sXT, SSD_STRB, wm + tm * 16, ks);
#pragma unroll
        for (int tn = 0; tn < 4; ++tn)
          aS[tm][tn] = wmma_bf16(af, ldb_bf(sBcT, SSD_STRB, wn + tn * 16, ks),
                                 aS[tm][tn]);
      }
    }
    __syncthreads();   // everyone done reading old S

    // ---- 5. state update: S = g*S + states ----
#pragma unroll
    for (int tm = 0; tm < 2; ++tm)
#pragma unroll
      for (int tn = 0; tn < 4; ++tn)
#pragma unroll
        for (int i = 0; i < 8; ++i) {
          int p = wm + tm * 16 + hi * 8 + i;
          int n = wn + tn * 16 + (lane & 15);
          sS[p * SSD_STRS + n] = gch * sS[p * SSD_STRS + n] + aS[tm][tn][i];
        }

    // ---- 6. epilogue: y = (Yd + exp(cs)*Yo + D*x) * silu(z) -> bf16 ----
#pragma unroll
    for (int tm = 0; tm < 2; ++tm)
#pragma unroll
      for (int tn = 0; tn < 4; ++tn)
#pragma unroll
        for (int i = 0; i < 8; ++i) {
          int q = wm + tm * 16 + hi * 8 + i;
          int p = wn + tn * 16 + (lane & 15);
          float yv = aY[tm][tn][i] + __expf(cs[q]) * aYo[tm][tn][i];
          size_t pos = tok0 + q;
          float xv = xconv[pos * CONV_DIM + g * HEADDIM + p];
          float zv = zx[pos * PROJ_OUT + h * HEADDIM + p];
          float out = (yv + Dh * xv) * siluf(zv);
          yact[pos * D_INNER + h * HEADDIM + p] = (__bf16)out;
        }
    __syncthreads();   // tiles/cs reusable next chunk
  }
}

// ---------------- launcher ----------------
extern "C" void kernel_launch(void* const* d_in, const int* in_sizes, int n_in,
                              void* d_out, int out_size, void* d_ws, size_t ws_size,
                              hipStream_t stream) {
  (void)in_sizes; (void)n_in; (void)out_size; (void)ws_size;

  const float* hidden  = (const float*)d_in[0];
  const float* w_in    = (const float*)d_in[1];
  const float* conv_w  = (const float*)d_in[2];
  const float* conv_b  = (const float*)d_in[3];
  const float* A_log   = (const float*)d_in[4];
  const float* Dp      = (const float*)d_in[5];
  const float* dt_bias = (const float*)d_in[6];
  const float* w_out   = (const float*)d_in[7];
  const float* ln1_w   = (const float*)d_in[8];
  const float* ln2_w   = (const float*)d_in[9];
  const float* gate_w  = (const float*)d_in[10];
  const float* up_w    = (const float*)d_in[11];
  const float* down_w  = (const float*)d_in[12];
  float* outp          = (float*)d_out;

  // workspace carving (256B aligned)
  char* ws = (char*)d_ws;
  size_t off = 0;
  auto take = [&](size_t bytes) {
    char* p = ws + off;
    off += (bytes + 255) & ~(size_t)255;
    return p;
  };
  float*  zxbcdt = (float*) take((size_t)TOK * PROJ_OUT * 4);   // 168 MB
  float*  xconv  = (float*) take((size_t)TOK * CONV_DIM * 4);   // 101 MB
  float*  dtbuf  = (float*) take((size_t)TOK * NHEADS * 4);
  float*  h1     = (float*) take((size_t)TOK * D_MODEL * 4);    // 67 MB
  __bf16* hn     = (__bf16*)take((size_t)TOK * D_MODEL * 2);
  __bf16* h2n    = (__bf16*)take((size_t)TOK * D_MODEL * 2);
  __bf16* yact   = (__bf16*)take((size_t)TOK * D_INNER * 2);
  __bf16* upbuf  = (__bf16*)take((size_t)TOK * INTER * 2);      // 117 MB
  __bf16* actmlp = (__bf16*)take((size_t)TOK * INTER * 2);      // 117 MB

  dim3 blk(256);

  // 1. h = rmsnorm(x, ln1) -> bf16
  rmsnorm_kernel<<<TOK, blk, 0, stream>>>(hidden, ln1_w, hn, D_MODEL);

  // 2. zxbcdt = h @ w_in^T  (f32 out)
  gemm_kernel<<<dim3(TOK / 128, (PROJ_OUT + 127) / 128), blk, 0, stream>>>(
      hn, w_in, TOK, PROJ_OUT, D_MODEL, zxbcdt, nullptr, nullptr, nullptr, 0);

  // 3. depthwise causal conv + silu over [x,B,C]
  conv_kernel<<<(TOK * CONV_DIM + 255) / 256, blk, 0, stream>>>(
      zxbcdt, conv_w, conv_b, xconv);

  // 4. dt = softplus(dt + bias)
  dt_kernel<<<(TOK * NHEADS + 255) / 256, blk, 0, stream>>>(
      zxbcdt, dt_bias, dtbuf);

  // 5. chunked SSD scan + gating -> yact (bf16)
  ssd_kernel<<<BB * NHEADS, blk, SSD_LDS_BYTES, stream>>>(
      xconv, dtbuf, zxbcdt, A_log, Dp, yact);

  // 6. h1 = hidden + yact @ w_out^T
  gemm_kernel<<<dim3(TOK / 128, D_MODEL / 128), blk, 0, stream>>>(
      yact, w_out, TOK, D_MODEL, D_INNER, h1, nullptr, hidden, nullptr, 1);

  // 7. h2 = rmsnorm(h1, ln2) -> bf16
  rmsnorm_kernel<<<TOK, blk, 0, stream>>>(h1, ln2_w, h2n, D_MODEL);

  // 8. up = h2 @ up_w^T (bf16 out)
  gemm_kernel<<<dim3(TOK / 128, INTER / 128), blk, 0, stream>>>(
      h2n, up_w, TOK, INTER, D_MODEL, nullptr, upbuf, nullptr, nullptr, 2);

  // 9. actmlp = silu(h2 @ gate_w^T) * up (bf16 out)
  gemm_kernel<<<dim3(TOK / 128, INTER / 128), blk, 0, stream>>>(
      h2n, gate_w, TOK, INTER, D_MODEL, nullptr, actmlp, nullptr, upbuf, 3);

  // 10. out = h1 + actmlp @ down_w^T
  gemm_kernel<<<dim3(TOK / 128, D_MODEL / 128), blk, 0, stream>>>(
      actmlp, down_w, TOK, D_MODEL, INTER, outp, nullptr, h1, nullptr, 1);
}
